// Attention_22316650070188
// MI455X (gfx1250) — matrix operand order repaired and checked by reading, not compiled, not run on hardware
//
#include <hip/hip_runtime.h>
#include <hip/hip_bf16.h>

// ---------------------------------------------------------------------------
// Shapes (fixed by the reference):
//   B=4, N=2048, C=1024, H=16, D=64, M=B*N=8192, 3C=3072
// ---------------------------------------------------------------------------
#define SB   4
#define SN   2048
#define SC   1024
#define SH   16
#define SD   64
#define SM   (SB * SN)     // 8192
#define S3C  (3 * SC)      // 3072

typedef __attribute__((ext_vector_type(16))) __bf16          v16bf;
typedef __attribute__((ext_vector_type(8)))  float           v8f;
typedef __attribute__((ext_vector_type(8)))  unsigned short  u16x8;
typedef __attribute__((ext_vector_type(4)))  unsigned int    u32x4;
typedef __attribute__((ext_vector_type(8)))  int             i32x8;
typedef __attribute__((ext_vector_type(4)))  int             i32x4;

union FragU {            // one WMMA 16-bit A/B operand = 2 x 16-byte chunks
    v16bf v;
    u16x8 h[2];
};

__device__ __forceinline__ unsigned short f2bf(float f) {
    unsigned int u = __float_as_uint(f);
    u += 0x7FFFu + ((u >> 16) & 1u);          // round-to-nearest-even
    return (unsigned short)(u >> 16);
}

__device__ __forceinline__ unsigned int lds_off(const void* p) {
    return (unsigned int)(unsigned long long)p;   // low 32 bits = LDS byte addr
}

// ---------------------------------------------------------------------------
// Tensor Data Mover: async 2-D tile load global -> LDS (D# per ISA 08).
//  cfg packs group1 dword0: data_size=2B, pad_enable, pad_interval/amount.
//  Row padding in LDS is expressed via the D# pad fields (16B pad per row).
// ---------------------------------------------------------------------------
#define TDM_CFG_ROW64   ((1u << 16) | (1u << 20) | (3u << 22) | (3u << 25)) // 64B rows + 16B pad
#define TDM_CFG_ROW128  ((1u << 16) | (1u << 20) | (4u << 22) | (3u << 25)) // 128B rows + 16B pad

__device__ __forceinline__ void tdm_load_2d(unsigned int lds_byte,
                                            const void* gptr,
                                            unsigned int cfg,
                                            unsigned int tile_d0,   // elems / row
                                            unsigned int tile_d1,   // rows
                                            unsigned long long row_stride_elems,
                                            unsigned int tensor_d0,
                                            unsigned int tensor_d1) {
    unsigned long long ga = (unsigned long long)gptr;
    u32x4 g0;
    g0.x = 1u;                                            // count=1 (user D#)
    g0.y = lds_byte;                                      // lds_addr
    g0.z = (unsigned int)ga;                              // global_addr[31:0]
    g0.w = ((unsigned int)(ga >> 32) & 0x01FFFFFFu) | (2u << 30); // addr[56:32] | type=2
    i32x8 g1;
    g1[0] = (int)cfg;                                     // mask=0,dsize,pad cfg
    g1[1] = (int)(tensor_d0 << 16);                       // tensor_dim0[15:0]
    g1[2] = (int)((tensor_d0 >> 16) | (tensor_d1 << 16)); // tdim0 hi | tdim1 lo
    g1[3] = (int)((tensor_d1 >> 16) | (tile_d0 << 16));   // tdim1 hi | tile_dim0
    g1[4] = (int)(tile_d1 & 0xFFFFu);                     // tile_dim1 (tile_dim2=0)
    g1[5] = (int)(unsigned int)row_stride_elems;          // dim0_stride[31:0]
    g1[6] = (int)((unsigned int)(row_stride_elems >> 32) & 0xFFFFu);
    g1[7] = 0;
    i32x4 z4 = {0, 0, 0, 0};
#if __has_include(<hip/amd_detail/amd_gfx1250_TDM.h>)
    i32x8 z8 = {0, 0, 0, 0, 0, 0, 0, 0};
    __builtin_amdgcn_tensor_load_to_lds(g0, g1, z4, z4, z8, 0);   // 6-arg toolchain
#else
    __builtin_amdgcn_tensor_load_to_lds(g0, g1, z4, z4, 0);       // ROCm 7.2 (5-arg)
#endif
}

// ---------------------------------------------------------------------------
// Kernel 1: fp32 -> bf16 bulk conversion
// ---------------------------------------------------------------------------
__global__ __launch_bounds__(256) void k_f32_to_bf16(const float* __restrict__ in,
                                                     unsigned short* __restrict__ out,
                                                     int n) {
    int i = blockIdx.x * 256 + threadIdx.x;
    if (i < n) out[i] = f2bf(in[i]);
}

// ---------------------------------------------------------------------------
// Kernel 2: bf16 GEMM, fp32 accumulate, TDM double-buffered K staging.
//   Cout[m, o] = sum_k A[m,k] * W[o,k] + bias[o]
// Block tile 128(M) x 128(N), 8 waves, each wave 64x32 (4x2 WMMA tiles).
// ---------------------------------------------------------------------------
__global__ __launch_bounds__(256) void k_gemm_bf16(const unsigned short* __restrict__ A,
                                                   const unsigned short* __restrict__ W,
                                                   const float* __restrict__ bias,
                                                   float* __restrict__ Cout,
                                                   int M, int Nn, int K) {
    __shared__ __align__(16) unsigned short As[2][128 * 40];   // 32 K + 8 pad
    __shared__ __align__(16) unsigned short Bs[2][128 * 40];

    const int tid  = threadIdx.x;
    const int wave = tid >> 5;
    const int lane = tid & 31;
    const int half = lane >> 4;
    const int ln   = lane & 15;
    const int wm   = wave >> 2;          // 0..1  -> 64 rows
    const int wn   = wave & 3;           // 0..3  -> 32 cols

    const int row0 = blockIdx.y * 128;
    const int col0 = blockIdx.x * 128;
    const int nsteps = K >> 5;

    v8f acc[4][2];
#pragma unroll
    for (int i = 0; i < 4; ++i)
#pragma unroll
        for (int n = 0; n < 2; ++n)
#pragma unroll
            for (int r = 0; r < 8; ++r)
                acc[i][n][r] = 0.0f;

    if (wave == 0) {   // prologue: DMA tile 0 into buffer 0
        tdm_load_2d(lds_off(&As[0][0]), &A[(size_t)row0 * K], TDM_CFG_ROW64,
                    32, 128, (unsigned long long)K, 32, 128);
        tdm_load_2d(lds_off(&Bs[0][0]), &W[(size_t)col0 * K], TDM_CFG_ROW64,
                    32, 128, (unsigned long long)K, 32, 128);
    }

    int buf = 0;
    for (int it = 0; it < nsteps; ++it, buf ^= 1) {
        if (wave == 0) {
            if (it + 1 < nsteps) {        // DMA next tile into other buffer
                int kb2 = (it + 1) << 5;
                tdm_load_2d(lds_off(&As[buf ^ 1][0]), &A[(size_t)row0 * K + kb2],
                            TDM_CFG_ROW64, 32, 128, (unsigned long long)K, 32, 128);
                tdm_load_2d(lds_off(&Bs[buf ^ 1][0]), &W[(size_t)col0 * K + kb2],
                            TDM_CFG_ROW64, 32, 128, (unsigned long long)K, 32, 128);
                __builtin_amdgcn_s_wait_tensorcnt(2);  // current tile complete
            } else {
                __builtin_amdgcn_s_wait_tensorcnt(0);
            }
        }
        __syncthreads();

        const unsigned short* as = As[buf];
        const unsigned short* bs = Bs[buf];

        // ---- B fragments (32x16, K = 16h..16h+15 contiguous per column) ----
        FragU bf[2];
#pragma unroll
        for (int n = 0; n < 2; ++n) {
            int col = wn * 32 + n * 16 + ln;
            bf[n].h[0] = *(const u16x8*)&bs[col * 40 + 16 * half];
            bf[n].h[1] = *(const u16x8*)&bs[col * 40 + 16 * half + 8];
        }

        // ---- A fragments + WMMA ----
#pragma unroll
        for (int i = 0; i < 4; ++i) {
            FragU af;
            int row = wm * 64 + i * 16 + ln;
            af.h[0] = *(const u16x8*)&as[row * 40 + 16 * half];       // K = 16h..16h+7
            af.h[1] = *(const u16x8*)&as[row * 40 + 16 * half + 8];   // K = 16h+8..16h+15
#pragma unroll
            for (int n = 0; n < 2; ++n)
                acc[i][n] = __builtin_amdgcn_wmma_f32_16x16x32_bf16(
                    false, af.v, false, bf[n].v, (short)0, acc[i][n], false, false);
        }
        __syncthreads();   // all reads done before TDM overwrites this buffer
    }

    // ---- epilogue: bias + fp32 store ----
#pragma unroll
    for (int i = 0; i < 4; ++i)
#pragma unroll
        for (int n = 0; n < 2; ++n) {
            int col = col0 + wn * 32 + n * 16 + ln;
            float bc = bias[col];
#pragma unroll
            for (int r = 0; r < 8; ++r) {
                int row = row0 + wm * 64 + i * 16 + r + 8 * half;
                Cout[(size_t)row * Nn + col] = acc[i][n][r] + bc;
            }
        }
}

// ---------------------------------------------------------------------------
// Kernel 3: per-(b,h,n) RMSNorm + RoPE for q,k; layout shuffle for v.
// One wave per row (D=64 -> 2 elements/lane).
//   q,k out: bf16 [B,H,N,D];   v out: bf16 [B,H,D,N] (transposed for PV)
// ---------------------------------------------------------------------------
__global__ __launch_bounds__(256) void k_normrope(const float* __restrict__ qkv, // [M,3C]
                                                  const float* __restrict__ qw,
                                                  const float* __restrict__ kw,
                                                  const float* __restrict__ cosT, // [N,D]
                                                  const float* __restrict__ sinT,
                                                  unsigned short* __restrict__ qbf,
                                                  unsigned short* __restrict__ kbf,
                                                  unsigned short* __restrict__ vT) {
    const int row_id = blockIdx.x * 8 + (threadIdx.x >> 5);
    const int lane   = threadIdx.x & 31;
    const int n = row_id & (SN - 1);
    const int h = (row_id >> 11) & (SH - 1);
    const int b = row_id >> 15;
    const size_t base = (size_t)((b << 11) | n) * S3C;
    const size_t bh   = (size_t)b * SH + h;

    const float c0 = cosT[n * SD + lane];
    const float s0 = sinT[n * SD + lane];
    const float c1 = cosT[n * SD + 32 + lane];
    const float s1 = sinT[n * SD + 32 + lane];

#pragma unroll
    for (int qk = 0; qk < 2; ++qk) {
        const float* src = qkv + base + qk * SC + h * SD;
        float x0 = src[lane], x1 = src[32 + lane];
        float ss = x0 * x0 + x1 * x1;
#pragma unroll
        for (int m = 16; m > 0; m >>= 1) ss += __shfl_xor(ss, m, 32);
        float rs = rsqrtf(ss * (1.0f / SD) + 1e-6f);
        const float* wv = qk ? kw : qw;
        float vn0 = x0 * rs * wv[lane];
        float vn1 = x1 * rs * wv[32 + lane];
        float y0 = vn0 * c0 - vn1 * s0;
        float y1 = vn1 * c1 + vn0 * s1;
        unsigned short* dst = (qk ? kbf : qbf) + ((bh * SN + n) * SD);
        dst[lane]      = f2bf(y0);
        dst[32 + lane] = f2bf(y1);
    }
    // v: transpose to [B,H,D,N]
    const float* vsrc = qkv + base + 2 * SC + h * SD;
    vT[(bh * SD + lane) * SN + n]        = f2bf(vsrc[lane]);
    vT[(bh * SD + 32 + lane) * SN + n]   = f2bf(vsrc[32 + lane]);
}

// ---------------------------------------------------------------------------
// Kernel 4: flash attention, one head per blockIdx.y, 128 queries per block.
// Each wave owns 16 queries; K/V^T tiles (64 keys) DMA'd by the TDM into
// double-buffered LDS; fp32 online softmax; P round-tripped through per-wave
// LDS to reach A-fragment layout for the PV WMMAs.
// ---------------------------------------------------------------------------
__global__ __launch_bounds__(256) void k_flash(const unsigned short* __restrict__ qbf,
                                               const unsigned short* __restrict__ kbf,
                                               const unsigned short* __restrict__ vT,
                                               unsigned short* __restrict__ obf) {
    __shared__ __align__(16) unsigned short Ks[2][64 * 72];     // [key][d] +8 pad
    __shared__ __align__(16) unsigned short Vs[2][64 * 72];     // [d][key] +8 pad
    __shared__ __align__(16) unsigned short Ps[8 * 16 * 64];    // per-wave 16x64 P

    const int tid  = threadIdx.x;
    const int wave = tid >> 5;
    const int lane = tid & 31;
    const int half = lane >> 4;
    const int ln   = lane & 15;

    const int bh = blockIdx.y;                 // b*H + h
    const int b  = bh >> 4;
    const int h  = bh & 15;
    const int q0 = blockIdx.x * 128 + wave * 16;

    // Q fragments, resident for the whole kernel (d = 0..31, 32..63)
    const size_t qbase = ((size_t)bh * SN + (q0 + ln)) * SD;
    FragU qf[2];
    qf[0].h[0] = *(const u16x8*)&qbf[qbase + 16 * half];
    qf[0].h[1] = *(const u16x8*)&qbf[qbase + 16 * half + 8];
    qf[1].h[0] = *(const u16x8*)&qbf[qbase + 32 + 16 * half];
    qf[1].h[1] = *(const u16x8*)&qbf[qbase + 32 + 16 * half + 8];

    v8f   o[4];
    float mrow[8], lrow[8];
#pragma unroll
    for (int dt = 0; dt < 4; ++dt)
#pragma unroll
        for (int r = 0; r < 8; ++r) o[dt][r] = 0.0f;
#pragma unroll
    for (int r = 0; r < 8; ++r) { mrow[r] = -1e30f; lrow[r] = 0.0f; }

    const unsigned short* kbase = kbf + (size_t)bh * SN * SD;  // [key][d]
    const unsigned short* vbase = vT + (size_t)bh * SD * SN;   // [d][key]

    if (wave == 0) {   // prologue: DMA key-tile 0 into buffer 0
        tdm_load_2d(lds_off(&Ks[0][0]), kbase, TDM_CFG_ROW128, 64, 64, 64, 64, 64);
        tdm_load_2d(lds_off(&Vs[0][0]), vbase, TDM_CFG_ROW128, 64, 64, SN, 64, 64);
    }

    const int nsteps = SN / 64;
    int buf = 0;
    for (int it = 0; it < nsteps; ++it, buf ^= 1) {
        if (wave == 0) {
            if (it + 1 < nsteps) {
                int k0n = (it + 1) * 64;
                tdm_load_2d(lds_off(&Ks[buf ^ 1][0]), kbase + (size_t)k0n * SD,
                            TDM_CFG_ROW128, 64, 64, 64, 64, 64);
                tdm_load_2d(lds_off(&Vs[buf ^ 1][0]), vbase + k0n,
                            TDM_CFG_ROW128, 64, 64, SN, 64, 64);
                __builtin_amdgcn_s_wait_tensorcnt(2);
            } else {
                __builtin_amdgcn_s_wait_tensorcnt(0);
            }
        }
        __syncthreads();

        const unsigned short* ks = Ks[buf];
        const unsigned short* vs = Vs[buf];

        // ---- S = Q K^T  (4 key subtiles x 2 K-steps) ----
        v8f s[4];
#pragma unroll
        for (int j = 0; j < 4; ++j) {
#pragma unroll
            for (int r = 0; r < 8; ++r) s[j][r] = 0.0f;
            FragU kf0, kf1;
            int key = j * 16 + ln;
            kf0.h[0] = *(const u16x8*)&ks[key * 72 + 16 * half];
            kf0.h[1] = *(const u16x8*)&ks[key * 72 + 16 * half + 8];
            kf1.h[0] = *(const u16x8*)&ks[key * 72 + 32 + 16 * half];
            kf1.h[1] = *(const u16x8*)&ks[key * 72 + 32 + 16 * half + 8];
            s[j] = __builtin_amdgcn_wmma_f32_16x16x32_bf16(
                false, qf[0].v, false, kf0.v, (short)0, s[j], false, false);
            s[j] = __builtin_amdgcn_wmma_f32_16x16x32_bf16(
                false, qf[1].v, false, kf1.v, (short)0, s[j], false, false);
        }

        // ---- online softmax (fp32); rows live as r+8*half across 16 lanes ----
#pragma unroll
        for (int r = 0; r < 8; ++r) {
            float a0 = s[0][r] * 0.125f, a1 = s[1][r] * 0.125f;
            float a2 = s[2][r] * 0.125f, a3 = s[3][r] * 0.125f;
            float mx = fmaxf(fmaxf(a0, a1), fmaxf(a2, a3));
#pragma unroll
            for (int m = 8; m > 0; m >>= 1) mx = fmaxf(mx, __shfl_xor(mx, m, 16));
            float mnew  = fmaxf(mrow[r], mx);
            float alpha = __expf(mrow[r] - mnew);
            float p0 = __expf(a0 - mnew), p1 = __expf(a1 - mnew);
            float p2 = __expf(a2 - mnew), p3 = __expf(a3 - mnew);
            float rsum = p0 + p1 + p2 + p3;
#pragma unroll
            for (int m = 8; m > 0; m >>= 1) rsum += __shfl_xor(rsum, m, 16);
            lrow[r] = lrow[r] * alpha + rsum;
            mrow[r] = mnew;
            s[0][r] = p0; s[1][r] = p1; s[2][r] = p2; s[3][r] = p3;
#pragma unroll
            for (int dt = 0; dt < 4; ++dt) o[dt][r] *= alpha;
        }

        // ---- P (C-layout) -> LDS bf16 -> A-fragment layout ----
        const int pw = wave * 1024;
#pragma unroll
        for (int j = 0; j < 4; ++j)
#pragma unroll
            for (int r = 0; r < 8; ++r)
                Ps[pw + (r + 8 * half) * 64 + j * 16 + ln] = f2bf(s[j][r]);
        __syncthreads();

        FragU pa0, pa1;
        pa0.h[0] = *(const u16x8*)&Ps[pw + ln * 64 + 16 * half];
        pa0.h[1] = *(const u16x8*)&Ps[pw + ln * 64 + 16 * half + 8];
        pa1.h[0] = *(const u16x8*)&Ps[pw + ln * 64 + 32 + 16 * half];
        pa1.h[1] = *(const u16x8*)&Ps[pw + ln * 64 + 32 + 16 * half + 8];

        // ---- O += P V  (4 d subtiles x 2 key-steps) ----
#pragma unroll
        for (int dt = 0; dt < 4; ++dt) {
            FragU vf0, vf1;
            int d = dt * 16 + ln;
            vf0.h[0] = *(const u16x8*)&vs[d * 72 + 16 * half];
            vf0.h[1] = *(const u16x8*)&vs[d * 72 + 16 * half + 8];
            vf1.h[0] = *(const u16x8*)&vs[d * 72 + 32 + 16 * half];
            vf1.h[1] = *(const u16x8*)&vs[d * 72 + 32 + 16 * half + 8];
            o[dt] = __builtin_amdgcn_wmma_f32_16x16x32_bf16(
                false, pa0.v, false, vf0.v, (short)0, o[dt], false, false);
            o[dt] = __builtin_amdgcn_wmma_f32_16x16x32_bf16(
                false, pa1.v, false, vf1.v, (short)0, o[dt], false, false);
        }
        __syncthreads();   // all reads done before TDM overwrites this buffer
    }

    // ---- finalize: O/l, write bf16 [B,N,C] with c = h*64 + d ----
#pragma unroll
    for (int dt = 0; dt < 4; ++dt) {
        int col = h * SD + dt * 16 + ln;
#pragma unroll
        for (int r = 0; r < 8; ++r) {
            int row = q0 + r + 8 * half;
            obf[((size_t)b * SN + row) * SC + col] = f2bf(o[dt][r] / lrow[r]);
        }
    }
}

// ---------------------------------------------------------------------------
// Launch
// ---------------------------------------------------------------------------
extern "C" void kernel_launch(void* const* d_in, const int* in_sizes, int n_in,
                              void* d_out, int out_size, void* d_ws, size_t ws_size,
                              hipStream_t stream) {
    const float* x      = (const float*)d_in[0];
    const float* qkv_w  = (const float*)d_in[1];
    const float* qkv_b  = (const float*)d_in[2];
    const float* proj_w = (const float*)d_in[3];
    const float* proj_b = (const float*)d_in[4];
    const float* q_nw   = (const float*)d_in[5];
    const float* k_nw   = (const float*)d_in[6];
    const float* cosT   = (const float*)d_in[7];
    const float* sinT   = (const float*)d_in[8];

    char* ws = (char*)d_ws;
    size_t off = 0;
    auto carve = [&](size_t bytes) -> char* {
        char* p = ws + off;
        off += (bytes + 255) & ~(size_t)255;
        return p;
    };
    unsigned short* x_bf    = (unsigned short*)carve((size_t)SM * SC * 2);
    unsigned short* w1_bf   = (unsigned short*)carve((size_t)S3C * SC * 2);
    unsigned short* w2_bf   = (unsigned short*)carve((size_t)SC * SC * 2);
    float*          qkv_f   = (float*)carve((size_t)SM * S3C * 4);
    unsigned short* q_bf    = (unsigned short*)carve((size_t)SB * SH * SN * SD * 2);
    unsigned short* k_bf    = (unsigned short*)carve((size_t)SB * SH * SN * SD * 2);
    unsigned short* vT_bf   = (unsigned short*)carve((size_t)SB * SH * SD * SN * 2);
    unsigned short* ao_bf   = (unsigned short*)carve((size_t)SM * SC * 2);

    // 1) fp32 -> bf16
    k_f32_to_bf16<<<(SM * SC + 255) / 256, 256, 0, stream>>>(x, x_bf, SM * SC);
    k_f32_to_bf16<<<(S3C * SC + 255) / 256, 256, 0, stream>>>(qkv_w, w1_bf, S3C * SC);
    k_f32_to_bf16<<<(SC * SC + 255) / 256, 256, 0, stream>>>(proj_w, w2_bf, SC * SC);

    // 2) QKV GEMM: [8192,1024] x [3072,1024]^T -> [8192,3072] fp32
    k_gemm_bf16<<<dim3(S3C / 128, SM / 128), 256, 0, stream>>>(
        x_bf, w1_bf, qkv_b, qkv_f, SM, S3C, SC);

    // 3) RMSNorm + RoPE + layout (one wave per (b,h,n))
    k_normrope<<<(SB * SH * SN) / 8, 256, 0, stream>>>(
        qkv_f, q_nw, k_nw, cosT, sinT, q_bf, k_bf, vT_bf);

    // 4) flash attention
    k_flash<<<dim3(SN / 128, SB * SH), 256, 0, stream>>>(q_bf, k_bf, vT_bf, ao_bf);

    // 5) output projection: [8192,1024] x [1024,1024]^T -> d_out fp32
    k_gemm_bf16<<<dim3(SC / 128, SM / 128), 256, 0, stream>>>(
        ao_bf, w2_bf, proj_b, (float*)d_out, SM, SC, SC);
}
